// PredictiveRnn_68917045231649
// MI455X (gfx1250) — matrix operand order repaired and checked by reading, not compiled, not run on hardware
//
#include <hip/hip_runtime.h>

typedef __attribute__((ext_vector_type(16))) _Float16 v16h;
typedef __attribute__((ext_vector_type(8)))  _Float16 v8h;
typedef __attribute__((ext_vector_type(8)))  float    v8f;
typedef __attribute__((ext_vector_type(4)))  float    v4f;

#define BATCH 128
#define TLEN  2048
#define HID   128
#define G3    384          // 3*HID
#define KAUG  160          // 128 hidden + [s, x_shift, 1, 0...] padding to 5 K-chunks
#define HSTR  168          // padded f16 A-buffer row stride (halves)
#define F32S  136          // padded f32 h row stride (floats)
#define ACCW  388          // padded acc row stride (floats)

// ---------------------------------------------------------------------------
// Prepass: pack augmented B matrix, f16, N-major:
//   bpack[n][k], k<128        = Wh[k][n]
//   bpack[n][128]             = n<256 ? Wi[0][n] : 0
//   bpack[n][129]             = n<256 ? Wi[1][n] : 0
//   bpack[n][130]             = n<256 ? bi[n]    : bhn[n-256]
//   bpack[n][131..159]        = 0
// ---------------------------------------------------------------------------
__global__ void prnn_pack_b(const float* __restrict__ Wh,
                            const float* __restrict__ Wi,
                            const float* __restrict__ bi,
                            const float* __restrict__ bhn,
                            _Float16* __restrict__ bpack) {
    int idx = blockIdx.x * 256 + threadIdx.x;
    if (idx >= G3 * KAUG) return;
    int n = idx / KAUG;
    int k = idx - n * KAUG;
    float v = 0.0f;
    if (k < HID)            v = Wh[k * G3 + n];
    else if (k == HID)      v = (n < 2 * HID) ? Wi[n]      : 0.0f;
    else if (k == HID + 1)  v = (n < 2 * HID) ? Wi[G3 + n] : 0.0f;
    else if (k == HID + 2)  v = (n < 2 * HID) ? bi[n]      : bhn[n - 2 * HID];
    bpack[n * KAUG + k] = (_Float16)v;
}

// ---------------------------------------------------------------------------
// Main recurrent kernel: 8 workgroups x 256 threads, 16 batch rows per WG.
// ---------------------------------------------------------------------------
__global__ __launch_bounds__(256) void prnn_scan_kernel(
    const float* __restrict__ s,  const float* __restrict__ x,
    const float* __restrict__ Wi, const float* __restrict__ bi,
    const _Float16* __restrict__ bpack,
    const float* __restrict__ Wd, const float* __restrict__ bd,
    float* __restrict__ out)
{
    __shared__ float    acc [16 * ACCW];   // 16 x 384 pre-activations (padded)
    __shared__ float    hf32[16 * F32S];   // h carry, f32
    __shared__ _Float16 hf16[16 * HSTR];   // augmented A operand [h | s x 1 0..]
    __shared__ float    part[256 * 2];     // Wd partial sums

    const int tid   = threadIdx.x;
    const int wave  = tid >> 5;
    const int lane  = tid & 31;
    const int bbase = blockIdx.x * 16;

    // ---- B fragments (augmented Wh), resident in VGPRs for the whole scan --
    v16h bfr[3][5];
    {
        const int nl = lane & 15;
        const int kb = (lane >> 4) * 16;
        #pragma unroll
        for (int ti = 0; ti < 3; ++ti) {
            int n = wave * 48 + ti * 16 + nl;
            #pragma unroll
            for (int kc = 0; kc < 5; ++kc)
                bfr[ti][kc] = *(const v16h*)(bpack + (size_t)n * KAUG + kc * 32 + kb);
        }
    }

    // ---- per-thread constants: only the n-gate input proj + readout -------
    const int gj = tid & 15;        // column group (8 cols each)
    const int gb = tid >> 4;        // local batch row 0..15
    const int c0 = gj * 8;
    float wi0n[8], wi1n[8], binn[8], wd0[8], wd1[8];
    #pragma unroll
    for (int e = 0; e < 8; ++e) {
        int col  = 2 * HID + c0 + e;
        wi0n[e] = Wi[col];
        wi1n[e] = Wi[G3 + col];
        binn[e] = bi[col];
        wd0[e]  = Wd[(c0 + e) * 2 + 0];
        wd1[e]  = Wd[(c0 + e) * 2 + 1];
    }
    const float bd0 = bd[0], bd1 = bd[1];

    const float* srow  = s + (size_t)(bbase + gb) * TLEN;
    const float* xrow  = x + (size_t)(bbase + gb) * TLEN;
    const int    rb    = (tid < 16) ? tid : 0;            // reduction row
    const float* xrowR = x   + (size_t)(bbase + rb) * TLEN;
    float*       outR  = out + (size_t)(bbase + rb) * TLEN;

    // ---- init A buffer: h0 = 0, specials [128]=s0 [129]=0 [130]=1 ---------
    for (int i = tid; i < 16 * HSTR; i += 256) hf16[i] = (_Float16)0.0f;
    for (int i = tid; i < 16 * F32S; i += 256) hf32[i] = 0.0f;
    __syncthreads();
    if (gj == 0) {
        hf16[gb * HSTR + 128] = (_Float16)srow[0];
        hf16[gb * HSTR + 130] = (_Float16)1.0f;
    }
    __syncthreads();

    // A-fragment lane layout (16-bit A): row m = lane%16,
// chunks at K = kc*32 + half*16 and +8
const int am = lane & 15;
const int ag = (lane >> 4) * 16;

    float sv = srow[0];   // s_t   (step-t input, already in A buffer)
    float xs = 0.0f;      // x_{t-1}

    for (int t = 0; t < TLEN; ++t) {
        // prefetch next-step inputs; they land under the WMMAs
        float sv_next = (t + 1 < TLEN) ? srow[t + 1] : 0.0f;
        float x_cur   = xrow[t];
        float xt      = xrowR[t];

        // ---- phase A: acc = [h|s,x,1] @ Bpack  (v_wmma_f32_16x16x32_f16) --
        v16h afr[5];
        #pragma unroll
        for (int kc = 0; kc < 5; ++kc) {
            const _Float16* base = hf16 + am * HSTR + kc * 32 + ag;
            v8h a0 = *(const v8h*)(base);
            v8h a1 = *(const v8h*)(base + 8);
            afr[kc] = __builtin_shufflevector(a0, a1,
                        0,1,2,3,4,5,6,7,8,9,10,11,12,13,14,15);
        }
        #pragma unroll
        for (int ti = 0; ti < 3; ++ti) {
            v8f c = {};
            #pragma unroll
            for (int kc = 0; kc < 5; ++kc)
                c = __builtin_amdgcn_wmma_f32_16x16x32_f16(
                        false, afr[kc], false, bfr[ti][kc],
                        (short)0, c, false, false);
            int col = wave * 48 + ti * 16 + (lane & 15);
            int r0  = (lane >> 4) * 8;
            #pragma unroll
            for (int v = 0; v < 8; ++v)
                acc[(r0 + v) * ACCW + col] = c[v];
        }
        __syncthreads();

        // ---- phase B: gates (r/z/bhn already folded in) + h update --------
        {
            const float* arow = acc + gb * ACCW;
            float ar[8], az[8], an[8], ho[8], hn[8];
            *(v4f*)&ar[0] = *(const v4f*)(arow + c0);
            *(v4f*)&ar[4] = *(const v4f*)(arow + c0 + 4);
            *(v4f*)&az[0] = *(const v4f*)(arow + HID + c0);
            *(v4f*)&az[4] = *(const v4f*)(arow + HID + c0 + 4);
            *(v4f*)&an[0] = *(const v4f*)(arow + 2 * HID + c0);
            *(v4f*)&an[4] = *(const v4f*)(arow + 2 * HID + c0 + 4);
            *(v4f*)&ho[0] = *(const v4f*)(hf32 + gb * F32S + c0);
            *(v4f*)&ho[4] = *(const v4f*)(hf32 + gb * F32S + c0 + 4);

            float pm = 0.0f, pv = 0.0f;
            #pragma unroll
            for (int e = 0; e < 8; ++e) {
                float r   = 1.0f / (1.0f + __expf(-ar[e]));   // ir folded in
                float z   = 1.0f / (1.0f + __expf(-az[e]));   // iz folded in
                float inn = sv * wi0n[e] + xs * wi1n[e] + binn[e];
                float np  = inn + r * an[e];                  // bhn folded in
                float e2  = __expf(-2.0f * np);
                float n   = (1.0f - e2) / (1.0f + e2);
                float h   = (1.0f - z) * n + z * ho[e];
                hn[e] = h;
                pm += h * wd0[e];
                pv += h * wd1[e];
            }
            *(v4f*)(hf32 + gb * F32S + c0)     = *(v4f*)&hn[0];
            *(v4f*)(hf32 + gb * F32S + c0 + 4) = *(v4f*)&hn[4];
            v8h h16;
            #pragma unroll
            for (int e = 0; e < 8; ++e) h16[e] = (_Float16)hn[e];
            *(v8h*)(hf16 + gb * HSTR + c0) = h16;
            if (gj == 0) {   // next step's augmented inputs
                hf16[gb * HSTR + 128] = (_Float16)sv_next;
                hf16[gb * HSTR + 129] = (_Float16)x_cur;
            }
            part[tid * 2 + 0] = pm;
            part[tid * 2 + 1] = pv;
        }
        __syncthreads();

        // ---- phase C: readout + logp (overlaps next step's WMMAs) ---------
        if (tid < 16) {
            float m = bd0, lv = bd1;
            #pragma unroll
            for (int j = 0; j < 16; ++j) {
                m  += part[(tid * 16 + j) * 2 + 0];
                lv += part[(tid * 16 + j) * 2 + 1];
            }
            float d  = xt - m;
            float lp = -0.9189385332046727f - 0.5f * lv
                       - 0.5f * d * d * __expf(-lv);
            outR[t] = lp;
        }

        sv = sv_next;
        xs = x_cur;
    }
}

// ---------------------------------------------------------------------------
extern "C" void kernel_launch(void* const* d_in, const int* in_sizes, int n_in,
                              void* d_out, int out_size, void* d_ws, size_t ws_size,
                              hipStream_t stream) {
    const float* s   = (const float*)d_in[0];
    const float* x   = (const float*)d_in[1];
    const float* Wi  = (const float*)d_in[2];
    const float* bi  = (const float*)d_in[3];
    const float* Wh  = (const float*)d_in[4];
    const float* bhn = (const float*)d_in[5];
    const float* Wd  = (const float*)d_in[6];
    const float* bd  = (const float*)d_in[7];
    float* out = (float*)d_out;

    _Float16* bpack = (_Float16*)d_ws;   // 384*160*2 = 120 KB scratch

    prnn_pack_b<<<(G3 * KAUG + 255) / 256, 256, 0, stream>>>(Wh, Wi, bi, bhn, bpack);
    prnn_scan_kernel<<<BATCH / 16, 256, 0, stream>>>(
        s, x, Wi, bi, bpack, Wd, bd, out);
}
